// Seq2Seq_65094524339190
// MI455X (gfx1250) — matrix operand order repaired and checked by reading, not compiled, not run on hardware
//
#include <hip/hip_runtime.h>

// ---------------------------------------------------------------------------
// Seq2Seq LSTM (2-layer encoder + 2-layer autoregressive decoder) for gfx1250
//   - persistent kernels: 64 blocks x 16 batch rows, 256 threads (8 wave32)
//   - V_WMMA_F32_16X16X32_F16 for all gate GEMMs, weights pinned in VGPRs
//   - embedding folded into layer-0 weights (x @ (Wih0*We)^T)
// ---------------------------------------------------------------------------

typedef __attribute__((ext_vector_type(16))) _Float16 v16h;
typedef __attribute__((ext_vector_type(8)))  float    v8f;

#define SRC_LEN  256
#define TRG_LEN  256
#define BATCH    1024
#define HID      128
#define G4       512   // 4*HID gate columns
#define K0       160   // layer-0 K: 32 (padded 6-dim input) + 128 (h)
#define K1       256   // layer-1 K: 128 (x) + 128 (h)
#define ROWS     16    // batch rows per block == WMMA M
#define NTHREADS 256   // 8 wave32

union Frag {           // one WMMA 16-bit operand fragment (16 halfs / lane)
    v16h  v;
    uint4 q[2];
};
union Hx8 {            // 8 halfs packed for a single b128 LDS store
    _Float16 h[8];
    uint4    q;
};

__device__ __forceinline__ float sigm(float x) {
    return 1.0f / (1.0f + __expf(-x));
}
__device__ __forceinline__ float tanh_fast(float x) {
    return 1.0f - 2.0f / (__expf(2.0f * x) + 1.0f);
}

// ---------------------------------------------------------------------------
// Prep: build folded f16 weight images + fused biases for one (enc|dec) stack.
//   W0[512][160] = [ Wih0@We (6 cols, padded to 32) | Whh0 ]   (f16)
//   W1[512][256] = [ Wih1 | Whh1 ]                             (f16)
//   b0[512] = Wih0@be + bih0 + bhh0 ;  b1[512] = bih1 + bhh1   (f32)
// ---------------------------------------------------------------------------
__global__ void prep_kernel(const float* __restrict__ Wih0, const float* __restrict__ We,
                            const float* __restrict__ be,
                            const float* __restrict__ Whh0,
                            const float* __restrict__ bih0, const float* __restrict__ bhh0,
                            const float* __restrict__ Wih1, const float* __restrict__ Whh1,
                            const float* __restrict__ bih1, const float* __restrict__ bhh1,
                            _Float16* __restrict__ W0, _Float16* __restrict__ W1,
                            float* __restrict__ b0, float* __restrict__ b1) {
    int g = blockIdx.x * blockDim.x + threadIdx.x;
    if (g >= G4) return;
    for (int c = 0; c < 32; ++c) {
        float s = 0.0f;
        if (c < 6) {
            for (int e = 0; e < 32; ++e) s += Wih0[g * 32 + e] * We[e * 6 + c];
        }
        W0[g * K0 + c] = (_Float16)s;
    }
    for (int k = 0; k < HID; ++k) W0[g * K0 + 32 + k] = (_Float16)Whh0[g * HID + k];
    float bs = bih0[g] + bhh0[g];
    for (int e = 0; e < 32; ++e) bs += Wih0[g * 32 + e] * be[e];
    b0[g] = bs;
    for (int k = 0; k < HID; ++k) {
        W1[g * K1 + k]       = (_Float16)Wih1[g * HID + k];
        W1[g * K1 + 128 + k] = (_Float16)Whh1[g * HID + k];
    }
    b1[g] = bih1[g] + bhh1[g];
}

// Build an A fragment (16x32 f16, contiguous k order, same as B) from an LDS row-major buffer.
// lane<16 holds row (lane&15), K = {0..15}; lane>=16 holds {16..31}.
__device__ __forceinline__ Frag load_a(const _Float16* buf, int rowStride,
                                       int nrel, int khi, int kbase) {
    Frag a;
    const uint4* p = (const uint4*)(buf + nrel * rowStride + kbase + khi * 16);
    a.q[0] = p[0];   // halfs +0 .. +7
    a.q[1] = p[1];   // halfs +8 .. +15
    return a;
}

// ---------------------------------------------------------------------------
// Encoder: fused layer0+layer1 recurrence, 256 steps, final h/c -> workspace.
// ---------------------------------------------------------------------------
__global__ __launch_bounds__(NTHREADS, 1) void enc_kernel(
        const float* __restrict__ src,
        const _Float16* __restrict__ W0, const _Float16* __restrict__ W1,
        const float* __restrict__ b0, const float* __restrict__ b1,
        float* __restrict__ h0o, float* __restrict__ c0o,
        float* __restrict__ h1o, float* __restrict__ c1o) {
    __shared__ __align__(16) _Float16 xbuf[ROWS * 32];
    __shared__ __align__(16) _Float16 hbuf0[ROWS * HID];
    __shared__ __align__(16) _Float16 hbuf1[ROWS * HID];
    __shared__ __align__(16) float    gbuf[ROWS * G4];
    __shared__ float bb0[G4];
    __shared__ float bb1[G4];

    const int tid  = threadIdx.x;
    const int lane = tid & 31;
    const int wave = tid >> 5;
    const int nrel = lane & 15;
    const int khi  = lane >> 4;
    const int rowbase = blockIdx.x * ROWS;
    const int urow = tid >> 4;
    const int ucol = (tid & 15) * 8;

    for (int i = tid; i < G4; i += NTHREADS) { bb0[i] = b0[i]; bb1[i] = b1[i]; }
    for (int i = tid; i < ROWS * 32; i += NTHREADS) xbuf[i] = (_Float16)0.0f;
    for (int i = tid; i < ROWS * HID; i += NTHREADS) {
        hbuf0[i] = (_Float16)0.0f;
        hbuf1[i] = (_Float16)0.0f;
    }

    // Pin weight fragments in VGPRs (loop invariant): 4 N-tiles/wave.
    Frag bf0[4][5];
    Frag bf1[4][8];
#pragma unroll
    for (int tt = 0; tt < 4; ++tt) {
        const int g = wave * 64 + tt * 16 + nrel;
        const uint4* p0 = (const uint4*)(W0 + g * K0 + khi * 16);
#pragma unroll
        for (int kt = 0; kt < 5; ++kt) {
            bf0[tt][kt].q[0] = p0[kt * 4 + 0];
            bf0[tt][kt].q[1] = p0[kt * 4 + 1];
        }
        const uint4* p1 = (const uint4*)(W1 + g * K1 + khi * 16);
#pragma unroll
        for (int kt = 0; kt < 8; ++kt) {
            bf1[tt][kt].q[0] = p1[kt * 4 + 0];
            bf1[tt][kt].q[1] = p1[kt * 4 + 1];
        }
    }

    float c0r[8], c1r[8], h0f[8], h1f[8];
#pragma unroll
    for (int i = 0; i < 8; ++i) { c0r[i] = 0.0f; c1r[i] = 0.0f; h0f[i] = 0.0f; h1f[i] = 0.0f; }

    __syncthreads();

#pragma unroll 1
    for (int t = 0; t < SRC_LEN; ++t) {
        // stage x_t (raw 6-dim src; embedding folded into W0)
        if (tid < 96) {
            int r = tid / 6, c = tid - r * 6;
            const float* sp = src + (size_t)(rowbase + r) * SRC_LEN * 6 + t * 6 + c;
            xbuf[r * 32 + c] = (_Float16)(*sp);
            if (t + 1 < SRC_LEN) __builtin_prefetch(sp + 6, 0, 3);
        }
        __syncthreads();

        // ---- layer 0: gates = [x | h0] @ W0^T + b0 ----
        {
            v8f acc[4];
#pragma unroll
            for (int tt = 0; tt < 4; ++tt) {
                float bv = bb0[wave * 64 + tt * 16 + nrel];
#pragma unroll
                for (int i = 0; i < 8; ++i) acc[tt][i] = bv;
            }
#pragma unroll
            for (int kt = 0; kt < 5; ++kt) {
                Frag a = (kt == 0) ? load_a(xbuf, 32, nrel, khi, 0)
                                   : load_a(hbuf0, HID, nrel, khi, (kt - 1) * 32);
#pragma unroll
                for (int tt = 0; tt < 4; ++tt)
                    acc[tt] = __builtin_amdgcn_wmma_f32_16x16x32_f16(
                        false, a.v, false, bf0[tt][kt].v, (short)0, acc[tt], false, false);
            }
#pragma unroll
            for (int tt = 0; tt < 4; ++tt) {
                const int col = wave * 64 + tt * 16 + nrel;
#pragma unroll
                for (int i = 0; i < 8; ++i) gbuf[(i + khi * 8) * G4 + col] = acc[tt][i];
            }
        }
        __syncthreads();

        // layer-0 pointwise update (c in registers, h -> LDS f16)
        {
            Hx8 hx;
#pragma unroll
            for (int i = 0; i < 8; ++i) {
                const int u = ucol + i;
                float gi = sigm(gbuf[urow * G4 + u]);
                float gf = sigm(gbuf[urow * G4 + 128 + u]);
                float gg = tanh_fast(gbuf[urow * G4 + 256 + u]);
                float go = sigm(gbuf[urow * G4 + 384 + u]);
                c0r[i] = gf * c0r[i] + gi * gg;
                float h = go * tanh_fast(c0r[i]);
                h0f[i]  = h;
                hx.h[i] = (_Float16)h;
            }
            *(uint4*)(hbuf0 + urow * HID + ucol) = hx.q;
        }
        __syncthreads();

        // ---- layer 1: gates = [h0 | h1] @ W1^T + b1 ----
        {
            v8f acc[4];
#pragma unroll
            for (int tt = 0; tt < 4; ++tt) {
                float bv = bb1[wave * 64 + tt * 16 + nrel];
#pragma unroll
                for (int i = 0; i < 8; ++i) acc[tt][i] = bv;
            }
#pragma unroll
            for (int kt = 0; kt < 8; ++kt) {
                Frag a = (kt < 4) ? load_a(hbuf0, HID, nrel, khi, kt * 32)
                                  : load_a(hbuf1, HID, nrel, khi, (kt - 4) * 32);
#pragma unroll
                for (int tt = 0; tt < 4; ++tt)
                    acc[tt] = __builtin_amdgcn_wmma_f32_16x16x32_f16(
                        false, a.v, false, bf1[tt][kt].v, (short)0, acc[tt], false, false);
            }
#pragma unroll
            for (int tt = 0; tt < 4; ++tt) {
                const int col = wave * 64 + tt * 16 + nrel;
#pragma unroll
                for (int i = 0; i < 8; ++i) gbuf[(i + khi * 8) * G4 + col] = acc[tt][i];
            }
        }
        __syncthreads();

        {
            Hx8 hx;
#pragma unroll
            for (int i = 0; i < 8; ++i) {
                const int u = ucol + i;
                float gi = sigm(gbuf[urow * G4 + u]);
                float gf = sigm(gbuf[urow * G4 + 128 + u]);
                float gg = tanh_fast(gbuf[urow * G4 + 256 + u]);
                float go = sigm(gbuf[urow * G4 + 384 + u]);
                c1r[i] = gf * c1r[i] + gi * gg;
                float h = go * tanh_fast(c1r[i]);
                h1f[i]  = h;
                hx.h[i] = (_Float16)h;
            }
            *(uint4*)(hbuf1 + urow * HID + ucol) = hx.q;
        }
        __syncthreads();
    }

    // hand final states to the decoder
#pragma unroll
    for (int i = 0; i < 8; ++i) {
        size_t idx = (size_t)(rowbase + urow) * HID + ucol + i;
        h0o[idx] = h0f[i];
        c0o[idx] = c0r[i];
        h1o[idx] = h1f[i];
        c1o[idx] = c1r[i];
    }
}

// ---------------------------------------------------------------------------
// Decoder: same fused recurrence + 6-wide output head + autoregressive feedback.
// ---------------------------------------------------------------------------
__global__ __launch_bounds__(NTHREADS, 1) void dec_kernel(
        const _Float16* __restrict__ W0, const _Float16* __restrict__ W1,
        const float* __restrict__ b0, const float* __restrict__ b1,
        const float* __restrict__ Wout, const float* __restrict__ bout,
        const float* __restrict__ h0i, const float* __restrict__ c0i,
        const float* __restrict__ h1i, const float* __restrict__ c1i,
        float* __restrict__ preds) {
    __shared__ __align__(16) _Float16 xbuf[ROWS * 32];
    __shared__ __align__(16) _Float16 hbuf0[ROWS * HID];
    __shared__ __align__(16) _Float16 hbuf1[ROWS * HID];
    __shared__ __align__(16) float    hbuf1f[ROWS * HID];
    __shared__ __align__(16) float    gbuf[ROWS * G4];
    __shared__ float bb0[G4];
    __shared__ float bb1[G4];
    __shared__ float woutl[6 * HID];
    __shared__ float boutl[6];

    const int tid  = threadIdx.x;
    const int lane = tid & 31;
    const int wave = tid >> 5;
    const int nrel = lane & 15;
    const int khi  = lane >> 4;
    const int rowbase = blockIdx.x * ROWS;
    const int urow = tid >> 4;
    const int ucol = (tid & 15) * 8;

    for (int i = tid; i < G4; i += NTHREADS) { bb0[i] = b0[i]; bb1[i] = b1[i]; }
    for (int i = tid; i < ROWS * 32; i += NTHREADS) xbuf[i] = (_Float16)0.0f; // inp0 = 0
    for (int i = tid; i < 6 * HID; i += NTHREADS) woutl[i] = Wout[i];
    if (tid < 6) boutl[tid] = bout[tid];

    // carry in encoder final states
    float c0r[8], c1r[8];
    {
        Hx8 hx0, hx1;
#pragma unroll
        for (int i = 0; i < 8; ++i) {
            size_t idx = (size_t)(rowbase + urow) * HID + ucol + i;
            c0r[i] = c0i[idx];
            c1r[i] = c1i[idx];
            float h0v = h0i[idx];
            float h1v = h1i[idx];
            hx0.h[i] = (_Float16)h0v;
            hx1.h[i] = (_Float16)h1v;
            hbuf1f[urow * HID + ucol + i] = h1v;
        }
        *(uint4*)(hbuf0 + urow * HID + ucol) = hx0.q;
        *(uint4*)(hbuf1 + urow * HID + ucol) = hx1.q;
    }

    Frag bf0[4][5];
    Frag bf1[4][8];
#pragma unroll
    for (int tt = 0; tt < 4; ++tt) {
        const int g = wave * 64 + tt * 16 + nrel;
        const uint4* p0 = (const uint4*)(W0 + g * K0 + khi * 16);
#pragma unroll
        for (int kt = 0; kt < 5; ++kt) {
            bf0[tt][kt].q[0] = p0[kt * 4 + 0];
            bf0[tt][kt].q[1] = p0[kt * 4 + 1];
        }
        const uint4* p1 = (const uint4*)(W1 + g * K1 + khi * 16);
#pragma unroll
        for (int kt = 0; kt < 8; ++kt) {
            bf1[tt][kt].q[0] = p1[kt * 4 + 0];
            bf1[tt][kt].q[1] = p1[kt * 4 + 1];
        }
    }

    __syncthreads();

#pragma unroll 1
    for (int t = 0; t < TRG_LEN; ++t) {
        // ---- layer 0: gates = [inp | h0] @ W0^T + b0 ----
        {
            v8f acc[4];
#pragma unroll
            for (int tt = 0; tt < 4; ++tt) {
                float bv = bb0[wave * 64 + tt * 16 + nrel];
#pragma unroll
                for (int i = 0; i < 8; ++i) acc[tt][i] = bv;
            }
#pragma unroll
            for (int kt = 0; kt < 5; ++kt) {
                Frag a = (kt == 0) ? load_a(xbuf, 32, nrel, khi, 0)
                                   : load_a(hbuf0, HID, nrel, khi, (kt - 1) * 32);
#pragma unroll
                for (int tt = 0; tt < 4; ++tt)
                    acc[tt] = __builtin_amdgcn_wmma_f32_16x16x32_f16(
                        false, a.v, false, bf0[tt][kt].v, (short)0, acc[tt], false, false);
            }
#pragma unroll
            for (int tt = 0; tt < 4; ++tt) {
                const int col = wave * 64 + tt * 16 + nrel;
#pragma unroll
                for (int i = 0; i < 8; ++i) gbuf[(i + khi * 8) * G4 + col] = acc[tt][i];
            }
        }
        __syncthreads();

        {
            Hx8 hx;
#pragma unroll
            for (int i = 0; i < 8; ++i) {
                const int u = ucol + i;
                float gi = sigm(gbuf[urow * G4 + u]);
                float gf = sigm(gbuf[urow * G4 + 128 + u]);
                float gg = tanh_fast(gbuf[urow * G4 + 256 + u]);
                float go = sigm(gbuf[urow * G4 + 384 + u]);
                c0r[i] = gf * c0r[i] + gi * gg;
                hx.h[i] = (_Float16)(go * tanh_fast(c0r[i]));
            }
            *(uint4*)(hbuf0 + urow * HID + ucol) = hx.q;
        }
        __syncthreads();

        // ---- layer 1: gates = [h0 | h1] @ W1^T + b1 ----
        {
            v8f acc[4];
#pragma unroll
            for (int tt = 0; tt < 4; ++tt) {
                float bv = bb1[wave * 64 + tt * 16 + nrel];
#pragma unroll
                for (int i = 0; i < 8; ++i) acc[tt][i] = bv;
            }
#pragma unroll
            for (int kt = 0; kt < 8; ++kt) {
                Frag a = (kt < 4) ? load_a(hbuf0, HID, nrel, khi, kt * 32)
                                  : load_a(hbuf1, HID, nrel, khi, (kt - 4) * 32);
#pragma unroll
                for (int tt = 0; tt < 4; ++tt)
                    acc[tt] = __builtin_amdgcn_wmma_f32_16x16x32_f16(
                        false, a.v, false, bf1[tt][kt].v, (short)0, acc[tt], false, false);
            }
#pragma unroll
            for (int tt = 0; tt < 4; ++tt) {
                const int col = wave * 64 + tt * 16 + nrel;
#pragma unroll
                for (int i = 0; i < 8; ++i) gbuf[(i + khi * 8) * G4 + col] = acc[tt][i];
            }
        }
        __syncthreads();

        {
            Hx8 hx;
#pragma unroll
            for (int i = 0; i < 8; ++i) {
                const int u = ucol + i;
                float gi = sigm(gbuf[urow * G4 + u]);
                float gf = sigm(gbuf[urow * G4 + 128 + u]);
                float gg = tanh_fast(gbuf[urow * G4 + 256 + u]);
                float go = sigm(gbuf[urow * G4 + 384 + u]);
                c1r[i] = gf * c1r[i] + gi * gg;
                float h = go * tanh_fast(c1r[i]);
                hx.h[i] = (_Float16)h;
                hbuf1f[urow * HID + ucol + i] = h;
            }
            *(uint4*)(hbuf1 + urow * HID + ucol) = hx.q;
        }
        __syncthreads();

        // ---- output head: out = h1 @ Wout^T + bout ; pred = sigmoid(out[:,5]) ----
        if (tid < 96) {
            int r = tid / 6, c = tid - r * 6;
            float s = boutl[c];
            for (int k = 0; k < HID; ++k) s += woutl[c * HID + k] * hbuf1f[r * HID + k];
            xbuf[r * 32 + c] = (_Float16)s;           // feedback (pre-sigmoid, full vec)
            if (c == 5) preds[(size_t)(rowbase + r) * TRG_LEN + t] = sigm(s);
        }
        __syncthreads();
    }
}

// ---------------------------------------------------------------------------
extern "C" void kernel_launch(void* const* d_in, const int* in_sizes, int n_in,
                              void* d_out, int out_size, void* d_ws, size_t ws_size,
                              hipStream_t stream) {
    (void)in_sizes; (void)n_in; (void)out_size; (void)ws_size;

    const float* src      = (const float*)d_in[0];
    const float* enc_We   = (const float*)d_in[1];
    const float* enc_be   = (const float*)d_in[2];
    const float* enc_Wih0 = (const float*)d_in[3];
    const float* enc_Whh0 = (const float*)d_in[4];
    const float* enc_bih0 = (const float*)d_in[5];
    const float* enc_bhh0 = (const float*)d_in[6];
    const float* enc_Wih1 = (const float*)d_in[7];
    const float* enc_Whh1 = (const float*)d_in[8];
    const float* enc_bih1 = (const float*)d_in[9];
    const float* enc_bhh1 = (const float*)d_in[10];
    const float* dec_We   = (const float*)d_in[11];
    const float* dec_be   = (const float*)d_in[12];
    const float* dec_Wih0 = (const float*)d_in[13];
    const float* dec_Whh0 = (const float*)d_in[14];
    const float* dec_bih0 = (const float*)d_in[15];
    const float* dec_bhh0 = (const float*)d_in[16];
    const float* dec_Wih1 = (const float*)d_in[17];
    const float* dec_Whh1 = (const float*)d_in[18];
    const float* dec_bih1 = (const float*)d_in[19];
    const float* dec_bhh1 = (const float*)d_in[20];
    const float* dec_Wout = (const float*)d_in[21];
    const float* dec_bout = (const float*)d_in[22];

    char* ws = (char*)d_ws;
    size_t off = 0;
    auto salloc = [&](size_t bytes) -> void* {
        void* p = ws + off;
        off = (off + bytes + 255) & ~(size_t)255;
        return p;
    };
    _Float16* W0e = (_Float16*)salloc((size_t)G4 * K0 * 2);
    _Float16* W1e = (_Float16*)salloc((size_t)G4 * K1 * 2);
    _Float16* W0d = (_Float16*)salloc((size_t)G4 * K0 * 2);
    _Float16* W1d = (_Float16*)salloc((size_t)G4 * K1 * 2);
    float* b0e = (float*)salloc(G4 * 4);
    float* b1e = (float*)salloc(G4 * 4);
    float* b0d = (float*)salloc(G4 * 4);
    float* b1d = (float*)salloc(G4 * 4);
    float* h0  = (float*)salloc((size_t)BATCH * HID * 4);
    float* c0  = (float*)salloc((size_t)BATCH * HID * 4);
    float* h1  = (float*)salloc((size_t)BATCH * HID * 4);
    float* c1  = (float*)salloc((size_t)BATCH * HID * 4);

    prep_kernel<<<2, 256, 0, stream>>>(enc_Wih0, enc_We, enc_be, enc_Whh0,
                                       enc_bih0, enc_bhh0, enc_Wih1, enc_Whh1,
                                       enc_bih1, enc_bhh1, W0e, W1e, b0e, b1e);
    prep_kernel<<<2, 256, 0, stream>>>(dec_Wih0, dec_We, dec_be, dec_Whh0,
                                       dec_bih0, dec_bhh0, dec_Wih1, dec_Whh1,
                                       dec_bih1, dec_bhh1, W0d, W1d, b0d, b1d);
    enc_kernel<<<BATCH / ROWS, NTHREADS, 0, stream>>>(src, W0e, W1e, b0e, b1e,
                                                      h0, c0, h1, c1);
    dec_kernel<<<BATCH / ROWS, NTHREADS, 0, stream>>>(W0d, W1d, b0d, b1d,
                                                      dec_Wout, dec_bout,
                                                      h0, c0, h1, c1,
                                                      (float*)d_out);
}
